// CrossMamba_41583873360038
// MI455X (gfx1250) — matrix operand order repaired and checked by reading, not compiled, not run on hardware
//
#include <hip/hip_runtime.h>

#define Bb      4
#define Hh      48
#define Ww      48
#define LL      (Hh*Ww)        // 2304
#define BL      (Bb*LL)        // 9216
#define DMODEL  128
#define DSSM    256
#define DSTATE  64
#define DBCC    132            // 2*64 + 4
#define BCROW   136            // padded row (16B-aligned rows of f16)
#define CHUNKQ  256
#define NCHUNK  9              // 2304/256

typedef __attribute__((ext_vector_type(16))) _Float16 v16h;
typedef __attribute__((ext_vector_type(8)))  _Float16 v8h;
typedef __attribute__((ext_vector_type(8)))  float    v8f;

__device__ __forceinline__ float sigmoidf_(float x){ return 1.f/(1.f+__expf(-x)); }
__device__ __forceinline__ float siluf_(float x){ return x*sigmoidf_(x); }
__device__ __forceinline__ float softplusf_(float x){ return (x>20.f)? x : log1pf(__expf(x)); }

// scan position q (direction dir) -> row-major spatial index h*W+w
__device__ __forceinline__ int spatial_idx(int dir, int q){
  int qq = (dir & 2) ? (LL-1-q) : q;
  if (dir & 1){ int w = qq / Hh; int h = qq % Hh; return h*Ww + w; }  // col-major scan
  return qq;
}

// ---- gfx1250 async global->LDS copy (ASYNCcnt-tracked), 16B per lane ----
__device__ __forceinline__ unsigned lds_off_u32(const void* p){
  return (unsigned)(size_t)p;   // low 32 bits of LDS-aperture flat address = LDS byte offset
}
__device__ __forceinline__ void async_b128(unsigned lds_off, const void* gaddr){
  asm volatile("global_load_async_to_lds_b128 %0, %1, off"
               :: "v"(lds_off), "v"(gaddr) : "memory");
}
__device__ __forceinline__ void wait_async(){
  asm volatile("s_wait_asynccnt 0x0" ::: "memory");
}

// ---- fragment-major LDS indexing (per-lane v16h register image) ----
// A-op 16x32 (MxK): lane = m + 16*((k>>3)&1); elem = (k&7) + 8*(k>>4)
__device__ __forceinline__ int a_frag_idx(int m, int k){
  return (m + 16*((k>>3)&1))*16 + (k&7) + 8*(k>>4);
}
// B-op 32x16 (KxN): lane = n + 16*((k>>3)&1); elem = (k&7) + 8*(k>>4)
__device__ __forceinline__ int b_frag_idx(int k, int n){
  return (n + 16*((k>>3)&1))*16 + (k&7) + 8*(k>>4);
}
__device__ __forceinline__ v16h load_frag_direct(const _Float16* F){
  return *(const v16h*)&F[(threadIdx.x & 31)*16];
}
// strided B-op load (for async-landed row-major [K][ldb] tiles)
__device__ __forceinline__ v16h load_b_strided(const _Float16* Bp, int ldb, int k0){
  int lane = threadIdx.x & 31; int n = lane & 15; int hf = lane >> 4;
  v16h b = {};
#pragma unroll
  for (int e=0;e<16;e++) b[e] = Bp[(k0 + 8*hf + (e&7) + 16*(e>>3))*ldb + n];
  return b;
}
__device__ __forceinline__ v8f wmma_f16(v16h a, v16h b, v8f c){
  return __builtin_amdgcn_wmma_f32_16x16x32_f16(false, a, false, b, (short)0, c, false, false);
}
// per-lane uniform row scaling of an A fragment (all 16 elems share one M row)
__device__ __forceinline__ v16h scale_frag(v16h a, float s){
  _Float16 hs = (_Float16)s;
#pragma unroll
  for (int e=0;e<16;e++) a[e] = a[e]*hs;
  return a;
}

// ---------------- generic GEMM: C[M,N] = A[M,K] @ W[K,N] (f32 in, f16 WMMA) -------------
__global__ void gemm_f16w(const float* __restrict__ A, const float* __restrict__ W,
                          float* __restrict__ C, int M, int N, int K){
  __shared__ __align__(32) _Float16 a_lds[512];   // fragment-major
  __shared__ __align__(32) _Float16 b_lds[512];   // fragment-major
  int tm = blockIdx.x, tn = blockIdx.y;
  int lane = threadIdx.x;
  v8f acc = {};
  for (int k0 = 0; k0 < K; k0 += 32){
    { // A panel: lane -> row r=lane>>1, 16 k's starting c0; land directly in frag layout
      int r = lane >> 1, c0 = (lane & 1) * 16;
      const float* Ar = &A[(size_t)(tm*16+r)*K + k0 + c0];
      float4 f0 = ((const float4*)Ar)[0];
      float4 f1 = ((const float4*)Ar)[1];
      float4 f2 = ((const float4*)Ar)[2];
      float4 f3 = ((const float4*)Ar)[3];
      if (k0 + 32 < K) __builtin_prefetch(Ar + 32, 0, 1);
      v8h h0 = {(_Float16)f0.x,(_Float16)f0.y,(_Float16)f0.z,(_Float16)f0.w,
                (_Float16)f1.x,(_Float16)f1.y,(_Float16)f1.z,(_Float16)f1.w};
      v8h h1 = {(_Float16)f2.x,(_Float16)f2.y,(_Float16)f2.z,(_Float16)f2.w,
                (_Float16)f3.x,(_Float16)f3.y,(_Float16)f3.z,(_Float16)f3.w};
      if (c0 == 0){
        *(v8h*)&a_lds[r*16]          = h0;  // k 0..7   -> lane r,    el 0..7
        *(v8h*)&a_lds[(r+16)*16]     = h1;  // k 8..15  -> lane r+16, el 0..7
      } else {
        *(v8h*)&a_lds[r*16 + 8]      = h0;  // k 16..23 -> lane r,    el 8..15
        *(v8h*)&a_lds[(r+16)*16 + 8] = h1;  // k 24..31 -> lane r+16, el 8..15
      }
    }
    { // B panel: lane -> row k0+lane, 16 cols; scatter into frag layout
      const float* Wr = &W[(size_t)(k0+lane)*N + tn*16];
      _Float16 hv[16];
      if (tn*16 + 16 <= N){
        float4 f0 = ((const float4*)Wr)[0];
        float4 f1 = ((const float4*)Wr)[1];
        float4 f2 = ((const float4*)Wr)[2];
        float4 f3 = ((const float4*)Wr)[3];
        if (k0 + 32 < K) __builtin_prefetch(Wr + (size_t)32*N, 0, 1);
        hv[0]=(_Float16)f0.x; hv[1]=(_Float16)f0.y; hv[2]=(_Float16)f0.z; hv[3]=(_Float16)f0.w;
        hv[4]=(_Float16)f1.x; hv[5]=(_Float16)f1.y; hv[6]=(_Float16)f1.z; hv[7]=(_Float16)f1.w;
        hv[8]=(_Float16)f2.x; hv[9]=(_Float16)f2.y; hv[10]=(_Float16)f2.z; hv[11]=(_Float16)f2.w;
        hv[12]=(_Float16)f3.x; hv[13]=(_Float16)f3.y; hv[14]=(_Float16)f3.z; hv[15]=(_Float16)f3.w;
      } else {
#pragma unroll
        for (int c=0;c<16;c++) hv[c] = (_Float16)((tn*16 + c < N) ? Wr[c] : 0.f);
      }
#pragma unroll
      for (int c=0;c<16;c++) b_lds[b_frag_idx(lane, c)] = hv[c];
    }
    __syncthreads();
    acc = wmma_f16(load_frag_direct(a_lds), load_frag_direct(b_lds), acc);
    __syncthreads();
  }
  int n = tn*16 + (lane & 15); int hf = lane >> 4;
  if (n < N){
#pragma unroll
    for (int r=0;r<8;r++){
      int m = tm*16 + r + 8*hf;
      C[(size_t)m*N + n] = acc[r];
    }
  }
}

// ------ depthwise 3x3 SAME conv + bias + SiLU; in [B][HW][C] f32 -> out [B][HW][rs] f16 ----
__global__ void dwconv_silu(const float* __restrict__ pre, const float* __restrict__ wgt,
                            const float* __restrict__ bias, _Float16* __restrict__ outp,
                            int C, int rs){
  int idx = blockIdx.x*blockDim.x + threadIdx.x;
  int total = Bb * C * LL;
  if (idx >= total) return;
  int l = idx % LL; int t = idx / LL; int c = t % C; int b = t / C;
  int h = l / Ww, w = l % Ww;
  float s = bias[c];
#pragma unroll
  for (int dy=-1; dy<=1; ++dy)
#pragma unroll
    for (int dx=-1; dx<=1; ++dx){
      int hh=h+dy, ww=w+dx;
      if (hh>=0 && hh<Hh && ww>=0 && ww<Ww)
        s += pre[((size_t)b*LL + hh*Ww+ww)*C + c] * wgt[c*9 + (dy+1)*3 + (dx+1)];
    }
  outp[((size_t)b*LL + l)*rs + c] = (_Float16)siluf_(s);
}

// ------------- SSD phase A: dt/cum/cdec + chunk states via WMMA ----------------
// block: 256 thr (8 waves); grid: B*4*9
__global__ void __launch_bounds__(256) ssd_states(
    const _Float16* __restrict__ bch, const _Float16* __restrict__ xsh,
    const float* __restrict__ dt_bias, const float* __restrict__ A_logs,
    float* __restrict__ cum_ws, float* __restrict__ dt_ws,
    float* __restrict__ states_ws, float* __restrict__ cdec_ws){
  __shared__ float dt_l[4][CHUNKQ];
  __shared__ float cum_l[4][CHUNKQ];
  __shared__ __align__(16) _Float16 xraw[32][256];   // raw x rows (async dest)
  __shared__ __align__(16) _Float16 braw[32][64];    // raw B rows (async dest)
  __shared__ __align__(32) _Float16 a_f[4][4][512];  // frag-major: w*x (M=p,K=q)
  __shared__ __align__(32) _Float16 b_f[4][512];     // frag-major: B (K=q,N=n)
  int bid = blockIdx.x;
  int chunk = bid % NCHUNK; int dir = (bid/NCHUNK)&3; int b = bid/(NCHUNK*4);
  int tid = threadIdx.x;
  int gh = dir*4;
  // softplus dt and dA
  for (int i = tid; i < 4*CHUNKQ; i += 256){
    int e = i >> 8; int q = i & 255;
    int sp = spatial_idx(dir, chunk*CHUNKQ + q);
    float raw = (float)bch[((size_t)b*LL + sp)*BCROW + 128 + e] + dt_bias[gh+e];
    float dtv = softplusf_(raw);
    dt_l[e][q] = dtv;
    cum_l[e][q] = dtv * (-__expf(A_logs[gh+e]));
  }
  __syncthreads();
  if (tid < 4){
    float s = 0.f;
    for (int q=0;q<CHUNKQ;q++){ s += cum_l[tid][q]; cum_l[tid][q] = s; }
  }
  __syncthreads();
  for (int i = tid; i < 4*CHUNKQ; i += 256){
    int e = i >> 8; int q = i & 255;
    size_t g = ((size_t)(b*4+dir)*LL + chunk*CHUNKQ + q)*4 + e;
    cum_ws[g] = cum_l[e][q]; dt_ws[g] = dt_l[e][q];
  }
  if (tid < 4) cdec_ws[((b*4+dir)*NCHUNK + chunk)*4 + tid] = __expf(cum_l[tid][CHUNKQ-1]);

  int wave = tid>>5; int lane = tid&31; int e_w = wave>>1; int ph = wave&1;
  v8f acc[8] = {};
  for (int t=0; t<8; ++t){
    int q0 = t*32;
    __syncthreads();
    // async-stage raw x rows (32 x 512B) and raw B rows (32 x 128B)
    for (int j = tid; j < 1024; j += 256){
      int kk = j >> 5, ln = j & 31;
      int sp = spatial_idx(dir, chunk*CHUNKQ + q0 + kk);
      async_b128(lds_off_u32(&xraw[kk][ln*8]),
                 xsh + ((size_t)b*LL + sp)*DSSM + ln*8);
    }
    {
      int kk = tid >> 3, l8 = tid & 7;
      int sp = spatial_idx(dir, chunk*CHUNKQ + q0 + kk);
      async_b128(lds_off_u32(&braw[kk][l8*8]),
                 bch + ((size_t)b*LL + sp)*BCROW + l8*8);
    }
    wait_async();
    __syncthreads();
    // build fragment-major operands from LDS
    for (int i = tid; i < 2048; i += 256){       // B: (K=kk, N=n)
      int kk = i & 31, n = i >> 5;
      b_f[n>>4][b_frag_idx(kk, n&15)] = braw[kk][n];
    }
    for (int i = tid; i < 8192; i += 256){       // A: w[q]*x[q,p] (M=p, K=q)
      int kk = i & 31; int p = (i>>5) & 63; int e = i >> 11;
      int q = q0 + kk;
      float wq = __expf(cum_l[e][CHUNKQ-1] - cum_l[e][q]) * dt_l[e][q];
      a_f[e][p>>4][a_frag_idx(p&15, kk)] = (_Float16)(wq * (float)xraw[kk][e*64+p]);
    }
    __syncthreads();
#pragma unroll
    for (int pi=0; pi<2; ++pi){
      v16h af = load_frag_direct(a_f[e_w][2*ph+pi]);
#pragma unroll
      for (int nt=0; nt<4; ++nt){
        acc[pi*4+nt] = wmma_f16(af, load_frag_direct(b_f[nt]), acc[pi*4+nt]);
      }
    }
  }
  float* sbase = states_ws + (size_t)(((b*4+dir)*NCHUNK + chunk)*4 + e_w)*64*64;
#pragma unroll
  for (int pi=0;pi<2;++pi)
#pragma unroll
    for (int nt=0;nt<4;++nt){
      int n = nt*16 + (lane&15);
#pragma unroll
      for (int r=0;r<8;++r){
        int p = (2*ph+pi)*16 + r + 8*(lane>>4);
        sbase[p*64 + n] = acc[pi*4+nt][r];
      }
    }
}

// ------------- inter-chunk state scan (sequential over 9 chunks) -----------------
__global__ void ssd_scan(const float* __restrict__ states_ws, const float* __restrict__ cdec_ws,
                         float* __restrict__ prev_ws){
  int bd = blockIdx.x;           // b*4+dir
  int tid = threadIdx.x;         // 256
  for (int idx = tid; idx < 4*64*64; idx += 256){
    int e = idx >> 12;
    float carry = 0.f;
    for (int c = 0; c < NCHUNK; ++c){
      size_t g = (size_t)(bd*NCHUNK + c)*4*64*64 + idx;
      prev_ws[g] = carry;
      carry = carry * cdec_ws[(bd*NCHUNK + c)*4 + e] + states_ws[g];
    }
  }
}

// ------------- SSD phase C: intra-chunk attention + C@prev + D*x ------------------
// block: 512 thr (16 waves: wave = e*4 + ptile); grid: B*4*9
__global__ void __launch_bounds__(512) ssd_y(
    const _Float16* __restrict__ bch, const _Float16* __restrict__ xsh,
    const float* __restrict__ cum_ws, const float* __restrict__ dt_ws,
    const float* __restrict__ prev_ws, const float* __restrict__ Ds,
    float* __restrict__ y_all){
  __shared__ float    cum_l[4][CHUNKQ];
  __shared__ float    dt_l[4][CHUNKQ];
  __shared__ __align__(32) _Float16 prev_f[4][4][2][512]; // frag B-op (K=n, two 32-steps)
  __shared__ __align__(16) _Float16 cq_l[16*64];          // C tile rows (async dest)
  __shared__ __align__(32) _Float16 cq_f[2][512];         // frag A-op (K=n, two 32-steps)
  __shared__ __align__(16) _Float16 braw[32][64];         // raw B rows (async dest)
  __shared__ __align__(32) _Float16 bn_f[2][2][512];      // frag B-op for CB [ksub][ns]
  __shared__ __align__(16) _Float16 x_l[4][32*64];        // [e][kk][p] (async dest; B-op)
  __shared__ float    cb_l[16*32];
  __shared__ __align__(32) _Float16 att_f[4][512];        // frag A-op per head
  int bid = blockIdx.x;
  int chunk = bid % NCHUNK; int dir = (bid/NCHUNK)&3; int b = bid/(NCHUNK*4);
  int tid = threadIdx.x; int lane = tid & 31;
  int wave = tid >> 5; int e_w = wave >> 2; int pt_w = wave & 3;
  for (int i=tid;i<4*CHUNKQ;i+=512){
    int e=i>>8, q=i&255;
    size_t g = ((size_t)(b*4+dir)*LL + chunk*CHUNKQ + q)*4 + e;
    cum_l[e][q]=cum_ws[g]; dt_l[e][q]=dt_ws[g];
  }
  const float* pbase = prev_ws + (size_t)((b*4+dir)*NCHUNK + chunk)*4*64*64;
  for (int i=tid;i<4*64*64;i+=512){
    int n=i&63; int p=(i>>6)&63; int e=i>>12;
    prev_f[e][p>>4][n>>5][b_frag_idx(n&31, p&15)] = (_Float16)pbase[(e*64+p)*64+n];
  }
  __syncthreads();
  float Dv = Ds[dir*4 + e_w];
  for (int qt=0; qt<16; ++qt){
    int qbase = qt*16;
    __syncthreads();
    if (tid < 128){   // async-stage C tile: 16 rows x 128B
      int m = tid >> 3, l8 = tid & 7;
      int sp = spatial_idx(dir, chunk*CHUNKQ + qbase + m);
      async_b128(lds_off_u32(&cq_l[m*64 + l8*8]),
                 bch + ((size_t)b*LL + sp)*BCROW + 64 + l8*8);
    }
    wait_async();
    __syncthreads();
    for (int i=tid;i<1024;i+=512){   // C tile -> frag A-op (K=n)
      int n=i&63, m=i>>6;
      cq_f[n>>5][a_frag_idx(m, n&31)] = cq_l[m*64+n];
    }
    __syncthreads();
    // inter-chunk term: y += (C * exp(cum_q)) @ prev^T  (row scale per lane)
    v8f acc = {};
    {
      float rs = __expf(cum_l[e_w][qbase + (lane & 15)]);
      acc = wmma_f16(scale_frag(load_frag_direct(cq_f[0]), rs),
                     load_frag_direct(prev_f[e_w][pt_w][0]), acc);
      acc = wmma_f16(scale_frag(load_frag_direct(cq_f[1]), rs),
                     load_frag_direct(prev_f[e_w][pt_w][1]), acc);
    }
    for (int t=0; t<=(qt>>1); ++t){
      int k0 = t*32;
      __syncthreads();
      for (int j = tid; j < 1024; j += 512){  // async x slab: 32 rows x 512B -> [e][kk][p]
        int kk = j >> 5, ln = j & 31;
        int sp = spatial_idx(dir, chunk*CHUNKQ + k0 + kk);
        async_b128(lds_off_u32(&x_l[ln>>3][kk*64 + (ln&7)*8]),
                   xsh + ((size_t)b*LL + sp)*DSSM + ln*8);
      }
      if (tid < 256){                          // async raw B rows: 32 rows x 128B
        int kk = tid >> 3, l8 = tid & 7;
        int sp = spatial_idx(dir, chunk*CHUNKQ + k0 + kk);
        async_b128(lds_off_u32(&braw[kk][l8*8]),
                   bch + ((size_t)b*LL + sp)*BCROW + l8*8);
      }
      wait_async();
      __syncthreads();
      for (int i=tid;i<2048;i+=512){           // B rows -> frag B-op for CB (K=n, N=ksub)
        int kk=i&31; int n=i>>5;
        bn_f[kk>>4][n>>5][b_frag_idx(n&31, kk&15)] = braw[kk][n];
      }
      __syncthreads();
      if (wave < 2){                           // CB tile (16q x 32k), K=n=64
        v8f cbacc = {};
        cbacc = wmma_f16(load_frag_direct(cq_f[0]), load_frag_direct(bn_f[wave][0]), cbacc);
        cbacc = wmma_f16(load_frag_direct(cq_f[1]), load_frag_direct(bn_f[wave][1]), cbacc);
        int n = lane & 15; int hf = lane>>4;
#pragma unroll
        for (int r=0;r<8;++r) cb_l[(r+8*hf)*32 + wave*16 + n] = cbacc[r];
      }
      __syncthreads();
      for (int i=tid;i<4*16*32;i+=512){        // att = mask * exp(cum_q-cum_k) * CB * dt_k
        int kk=i&31; int m=(i>>5)&15; int e=i>>9;
        int q = qbase+m; int k = k0+kk;
        float v = 0.f;
        if (k <= q) v = __expf(cum_l[e][q]-cum_l[e][k]) * cb_l[m*32+kk] * dt_l[e][k];
        att_f[e][a_frag_idx(m, kk)] = (_Float16)v;
      }
      __syncthreads();
      acc = wmma_f16(load_frag_direct(att_f[e_w]),
                     load_b_strided(&x_l[e_w][pt_w*16], 64, 0), acc);
    }
    // store y tile (+ D*x skip)
    int p = pt_w*16 + (lane&15);
    int ch = e_w*64 + p;
#pragma unroll
    for (int r=0;r<8;++r){
      int q = qbase + r + 8*(lane>>4);
      int sp = spatial_idx(dir, chunk*CHUNKQ + q);
      float xv = (float)xsh[((size_t)b*LL + sp)*DSSM + ch];
      y_all[((size_t)((b*4+dir)*LL) + chunk*CHUNKQ + q)*DSSM + ch] = acc[r] + Dv*xv;
    }
  }
}

// ------------- merge 4 directions + gate + RMSNorm -------------------------------
__global__ void combine_norm(const float* __restrict__ y_all, const float* __restrict__ z,
                             const float* __restrict__ norm_w, float* __restrict__ t_out){
  __shared__ float red[256];
  int bl = blockIdx.x;               // b*L + l (row-major)
  int b = bl / LL; int l = bl % LL;
  int c = threadIdx.x;
  int h = l / Ww, w = l % Ww;
  int q1 = w*Hh + h;
  size_t base = (size_t)b*4*LL;
  float s = y_all[(base + 0*LL + l)        *DSSM + c]
          + y_all[(base + 2*LL + (LL-1-l)) *DSSM + c]
          + y_all[(base + 1*LL + q1)       *DSSM + c]
          + y_all[(base + 3*LL + (LL-1-q1))*DSSM + c];
  float zv = z[(size_t)bl*DSSM + c];
  float t = s * siluf_(zv);
  red[c] = t*t;
  __syncthreads();
  for (int o=128;o>0;o>>=1){ if (c<o) red[c]+=red[c+o]; __syncthreads(); }
  float r = rsqrtf(red[0]/(float)DSSM + 1e-5f);
  t_out[(size_t)bl*DSSM + c] = t * r * norm_w[c];
}

extern "C" void kernel_launch(void* const* d_in, const int* in_sizes, int n_in,
                              void* d_out, int out_size, void* d_ws, size_t ws_size,
                              hipStream_t stream) {
  (void)in_sizes; (void)n_in; (void)out_size; (void)ws_size;
  const float* u1      = (const float*)d_in[0];
  const float* u1cu2   = (const float*)d_in[2];
  const float* Wskip   = (const float*)d_in[4];
  const float* Wxs     = (const float*)d_in[5];
  const float* Wbcdt   = (const float*)d_in[6];
  const float* xs_cw   = (const float*)d_in[7];
  const float* xs_cb   = (const float*)d_in[8];
  const float* bc_cw   = (const float*)d_in[9];
  const float* bc_cb   = (const float*)d_in[10];
  const float* dt_bias = (const float*)d_in[11];
  const float* A_logs  = (const float*)d_in[12];
  const float* Ds      = (const float*)d_in[13];
  const float* norm_w  = (const float*)d_in[14];
  const float* Wout    = (const float*)d_in[15];

  float* ws = (float*)d_ws;
  size_t o = 0;
  float* z_buf   = ws + o;  o += (size_t)BL*DSSM;
  float* xs_pre  = ws + o;  o += (size_t)BL*DSSM;
  float* bc_pre  = ws + o;  o += (size_t)BL*DBCC;
  _Float16* xsh  = (_Float16*)(ws + o);  o += (size_t)BL*DSSM/2;   // f16 [b][sp][256]
  _Float16* bch  = (_Float16*)(ws + o);  o += (size_t)BL*BCROW/2;  // f16 [b][sp][136]
  float* cum_ws  = ws + o;  o += (size_t)Bb*4*LL*4;
  float* dt_ws   = ws + o;  o += (size_t)Bb*4*LL*4;
  float* states  = ws + o;  o += (size_t)Bb*4*NCHUNK*4*64*64;
  float* cdec    = ws + o;  o += (size_t)Bb*4*NCHUNK*4;
  float* prev    = ws + o;  o += (size_t)Bb*4*NCHUNK*4*64*64;
  float* y_all   = ws + o;  o += (size_t)Bb*4*LL*DSSM;
  float* t_norm  = xs_pre;  // xs_pre is dead after the conv; reuse

  dim3 gz(BL/16, DSSM/16);
  gemm_f16w<<<gz, 32, 0, stream>>>(u1, Wskip, z_buf, BL, DSSM, DMODEL);
  gemm_f16w<<<gz, 32, 0, stream>>>(u1, Wxs, xs_pre, BL, DSSM, DMODEL);
  dim3 gb(BL/16, (DBCC+15)/16);
  gemm_f16w<<<gb, 32, 0, stream>>>(u1cu2, Wbcdt, bc_pre, BL, DBCC, DMODEL);

  int tot1 = Bb*DSSM*LL;
  dwconv_silu<<<(tot1+255)/256, 256, 0, stream>>>(xs_pre, xs_cw, xs_cb, xsh, DSSM, DSSM);
  int tot2 = Bb*DBCC*LL;
  dwconv_silu<<<(tot2+255)/256, 256, 0, stream>>>(bc_pre, bc_cw, bc_cb, bch, DBCC, BCROW);

  ssd_states<<<Bb*4*NCHUNK, 256, 0, stream>>>(bch, xsh, dt_bias, A_logs,
                                              cum_ws, dt_ws, states, cdec);
  ssd_scan<<<Bb*4, 256, 0, stream>>>(states, cdec, prev);
  ssd_y<<<Bb*4*NCHUNK, 512, 0, stream>>>(bch, xsh, cum_ws, dt_ws, prev, Ds, y_all);

  combine_norm<<<BL, 256, 0, stream>>>(y_all, z_buf, norm_w, t_norm);

  dim3 go(BL/16, DMODEL/16);
  gemm_f16w<<<go, 32, 0, stream>>>(t_norm, Wout, (float*)d_out, BL, DMODEL, DSSM);
}
